// MixLoss_84507776516440
// MI455X (gfx1250) — matrix operand order repaired and checked by reading, not compiled, not run on hardware
//
#include <hip/hip_runtime.h>

typedef __attribute__((ext_vector_type(16))) _Float16 v16h;
typedef __attribute__((ext_vector_type(8)))  float    v8f;
typedef __attribute__((ext_vector_type(4)))  _Float16 h4;

#define B_  32
#define S_  2048
#define F_  1024
#define C_  64
#define NROWS (B_*S_)             // 65536 rows
#define ROWS_PER_BLK 16
#define NBLK (NROWS/ROWS_PER_BLK) // 4096 blocks
#define THREADS 128               // 4 waves (wave32)
#define NCHUNK (F_/32)            // 32 K-chunks of 32
#define EH_STRIDE 1032            // 1024 + 8 halfs pad (16B) to break bank conflicts

// Monotone float -> unsigned key so atomicMax(unsigned) implements float max.
__device__ __forceinline__ unsigned fkey(float x) {
    unsigned u = __float_as_uint(x);
    return (u & 0x80000000u) ? ~u : (u | 0x80000000u);
}
__device__ __forceinline__ float fromkey(unsigned k) {
    unsigned u = (k & 0x80000000u) ? (k & 0x7fffffffu) : ~k;
    return __uint_as_float(u);
}

// seg[f] = argmax_c mask[f][c]  (mask rows are one-hot)
__global__ void prep_seg(const float* __restrict__ mask, int* __restrict__ seg) {
    int f = blockIdx.x * blockDim.x + threadIdx.x;
    if (f >= F_) return;
    const float* row = mask + (size_t)f * C_;
    int best = 0; float bv = row[0];
    for (int c = 1; c < C_; ++c) { float v = row[c]; if (v > bv) { bv = v; best = c; } }
    seg[f] = best;
}

// Pre-pack mask into WMMA B-fragment order:
// bpack[((nt*NCHUNK + kc)*32 + lane)*16 + j] = mask[(K)*C_ + N] as f16, where
//   K = kc*32 + (lane<16 ? 0 : 16) + j   (k order used here, as for A: lanes 0-15 K 0-15, 16-31 K 16-31)
//   N = nt*16 + (lane & 15)
__global__ void prep_bpack(const float* __restrict__ mask, _Float16* __restrict__ bpack) {
    int t = blockIdx.x * blockDim.x + threadIdx.x;   // 0 .. 4*NCHUNK*32-1
    if (t >= 4 * NCHUNK * 32) return;
    int lane = t & 31;
    int kc   = (t >> 5) & (NCHUNK - 1);
    int nt   = t >> 10;
    int bkb  = kc * 32 + ((lane >> 4) << 4);
    int bcol = nt * 16 + (lane & 15);
    _Float16* dst = bpack + (size_t)t * 16;
    #pragma unroll
    for (int j = 0; j < 16; ++j)
        dst[j] = (_Float16)mask[(size_t)(bkb + j) * C_ + bcol];
}

__global__ __launch_bounds__(THREADS)
void mix_main(const float* __restrict__ logits,
              const int* __restrict__ labels,
              const int* __restrict__ seg,
              const _Float16* __restrict__ bpack,
              float* __restrict__ partials) {
    __shared__ _Float16 eh[ROWS_PER_BLK * EH_STRIDE];   // exp(x-m) tile, f16 (~33KB)
    __shared__ unsigned ckey[ROWS_PER_BLK * C_];        // per-row coarse-max keys (4KB)
    __shared__ float    Sld[ROWS_PER_BLK * C_];         // WMMA result: coarse exp-sums (4KB)
    __shared__ int      segL[F_];                       // cached fine->coarse map (4KB)
    __shared__ float    rowm[ROWS_PER_BLK];
    __shared__ float    rv[ROWS_PER_BLK];

    const int tid  = threadIdx.x;
    const int lane = tid & 31;
    const int wave = tid >> 5;          // 0..3

    // ---- init LDS state ----
    for (int i = tid; i < ROWS_PER_BLK * C_; i += THREADS) ckey[i] = 0u;  // key 0 == most negative
    for (int i = tid; i < F_; i += THREADS) segL[i] = seg[i];
    __syncthreads();

    const float* xb = logits + (size_t)blockIdx.x * (ROWS_PER_BLK * F_);

    // ---- pass 1: coalesced stream, coarse-segment max (row max falls out of the bins) ----
    for (int i = 0; i < (ROWS_PER_BLK * F_) / (THREADS * 4); ++i) {
        int g = (i * THREADS + tid) * 4;                // float index in tile
        const float4 x = *(const float4*)(xb + g);
        int r = g >> 10;                                // row within tile (F_=1024)
        int f = g & (F_ - 1);
        atomicMax(&ckey[r * C_ + segL[f + 0]], fkey(x.x));
        atomicMax(&ckey[r * C_ + segL[f + 1]], fkey(x.y));
        atomicMax(&ckey[r * C_ + segL[f + 2]], fkey(x.z));
        atomicMax(&ckey[r * C_ + segL[f + 3]], fkey(x.w));
    }
    __syncthreads();
    // row max = max over the 64 coarse bins (every fine class is in some bin)
    if (tid < ROWS_PER_BLK) {
        unsigned mk = 0u;
        for (int c = 0; c < C_; ++c) { unsigned k = ckey[tid * C_ + c]; if (k > mk) mk = k; }
        rowm[tid] = fromkey(mk);
    }
    __syncthreads();

    // ---- pass 2: e = exp(x - m) -> f16 LDS tile (re-read is WGP$/L2-hot) ----
    for (int i = 0; i < (ROWS_PER_BLK * F_) / (THREADS * 4); ++i) {
        int g = (i * THREADS + tid) * 4;
        const float4 x = *(const float4*)(xb + g);
        int r = g >> 10;
        int f = g & (F_ - 1);
        float m = rowm[r];
        h4 e;
        e[0] = (_Float16)__expf(x.x - m);
        e[1] = (_Float16)__expf(x.y - m);
        e[2] = (_Float16)__expf(x.z - m);
        e[3] = (_Float16)__expf(x.w - m);
        *(h4*)(&eh[r * EH_STRIDE + f]) = e;
    }
    __syncthreads();

    // ---- WMMA pooling: S[16x64] = E[16x1024] (f16) @ mask[1024x64] (f16), f32 acc ----
    {
        const int ntile = wave;                 // each wave owns 16 coarse columns
        const int arow  = lane & 15;            // A: M = lane % 16
        const int ahalf = lane >> 4;            // A k order used here: lanes 0-15 K 0-15, 16-31 K 16-31
        const _Float16* ar = &eh[arow * EH_STRIDE];
        // pre-packed B fragments: one contiguous 32B load per lane per chunk
        const v16h* bp = ((const v16h*)bpack) + ((size_t)ntile * NCHUNK * 32 + lane);
        v8f acc = {};
        #pragma unroll 4
        for (int kc = 0; kc < NCHUNK; ++kc) {
            v16h a;
            int kb = kc * 32 + ahalf * 16;
            #pragma unroll
            for (int j = 0; j < 8; ++j) { a[j] = ar[kb + j]; a[8 + j] = ar[kb + 8 + j]; }
            v16h b = bp[(size_t)kc * 32];
            acc = __builtin_amdgcn_wmma_f32_16x16x32_f16(
                /*neg_a=*/false, a, /*neg_b=*/false, b,
                /*c_mod=*/(short)0, acc, /*reuse_a=*/false, /*reuse_b=*/false);
        }
        // C/D layout: N = lane % 16, VGPR v holds M = v + (lane<16 ? 0 : 8)
        int n  = lane & 15;
        int mb = ahalf * 8;
        #pragma unroll
        for (int v = 0; v < 8; ++v) Sld[(mb + v) * C_ + ntile * 16 + n] = acc[v];
    }
    __syncthreads();

    // ---- per-row losses (Z comes free as the row-sum of S) ----
    if (tid < ROWS_PER_BLK) {
        int r = tid;
        int row = blockIdx.x * ROWS_PER_BLK + r;
        int lab = labels[row];
        float Z = 0.f;
        for (int c = 0; c < C_; ++c) Z += Sld[r * C_ + c];
        float nll = -(__logf(Sld[r * C_ + lab]) - __logf(Z));

        float M2 = rowm[r];                      // max over coarse bins == row max
        float lse = 0.f;
        for (int c = 0; c < C_; ++c) lse += __expf(fromkey(ckey[r * C_ + c]) - M2);
        float ce = -(fromkey(ckey[r * C_ + lab]) - M2 - __logf(lse));
        rv[r] = 0.5f * (ce + nll);
    }
    __syncthreads();
    if (tid == 0) {
        float s = 0.f;
        for (int r = 0; r < ROWS_PER_BLK; ++r) s += rv[r];
        partials[blockIdx.x] = s;
    }
}

__global__ void finalize_kernel(const float* __restrict__ partials, float* __restrict__ out) {
    __shared__ float red[256];
    float s = 0.f;
    for (int i = threadIdx.x; i < NBLK; i += 256) s += partials[i];
    red[threadIdx.x] = s;
    __syncthreads();
    for (int off = 128; off > 0; off >>= 1) {
        if ((int)threadIdx.x < off) red[threadIdx.x] += red[threadIdx.x + off];
        __syncthreads();
    }
    if (threadIdx.x == 0) out[0] = red[0] / (float)NROWS;
}

extern "C" void kernel_launch(void* const* d_in, const int* in_sizes, int n_in,
                              void* d_out, int out_size, void* d_ws, size_t ws_size,
                              hipStream_t stream) {
    (void)in_sizes; (void)n_in; (void)out_size; (void)ws_size;
    const float* logits = (const float*)d_in[0];
    const int*   labels = (const int*)d_in[1];     // integer input -> const int*
    const float* mask   = (const float*)d_in[2];
    float* out = (float*)d_out;

    // workspace layout: [partials: NBLK f32][seg: F_ i32][bpack: 4*NCHUNK*32*16 f16 (128KB)]
    float*    partials = (float*)d_ws;
    int*      seg      = (int*)(partials + NBLK);
    _Float16* bpack    = (_Float16*)(seg + F_);    // offset 20KB from ws base: 32B-aligned

    prep_seg<<<(F_ + 255) / 256, 256, 0, stream>>>(mask, seg);
    prep_bpack<<<(4 * NCHUNK * 32 + 255) / 256, 256, 0, stream>>>(mask, bpack);
    mix_main<<<NBLK, THREADS, 0, stream>>>(logits, labels, seg, bpack, partials);
    finalize_kernel<<<1, 256, 0, stream>>>(partials, out);
}
